// Attention_w_KVCache_38654705664082
// MI455X (gfx1250) — matrix operand order repaired and checked by reading, not compiled, not run on hardware
//
#include <hip/hip_runtime.h>
#include <math.h>

// ---------------------------------------------------------------------------
// Attention with KV cache, MI455X (gfx1250, wave32, WMMA + Tensor Data Mover).
//   1) converts: x -> bf16; w_qkv/w_proj -> bf16 TRANSPOSED [N][K];
//      K cache -> bf16 [B,H,5120,64]; V cache -> bf16 TRANSPOSED [B,H,64,5120]
//   2) QKV GEMM: bf16 WMMA, f32 acc, double-buffered TDM staging of A and B^T
//      tiles; epilogue fuses head split, Q*scale, K/V scatter
//   3) Flash attention, 64-key chunks, K and V tiles streamed by TDM while
//      WMMA computes the previous chunk; f32 online softmax
//   4) Proj GEMM (same TDM pipeline) fused with bias -> f32 d_out
// ---------------------------------------------------------------------------

#define B_  4
#define N_  1024
#define C_  1024
#define H_  16
#define D_  64
#define LC_ 4096
#define LT_ 5120   // LC_ + N_
#define CHUNK 64   // flash key-chunk

typedef __attribute__((ext_vector_type(16))) __bf16 v16bf;
typedef __attribute__((ext_vector_type(8)))  float  v8f;
typedef __attribute__((ext_vector_type(4)))  unsigned v4u;
typedef __attribute__((ext_vector_type(8)))  int      v8i;
typedef __attribute__((ext_vector_type(4)))  int      v4i;

// Tensor Data Mover availability (device pass; host parse pass -> fallback)
#ifdef __has_builtin
#if __has_builtin(__builtin_amdgcn_tensor_load_to_lds) && \
    __has_builtin(__builtin_amdgcn_s_wait_tensorcnt)
#define USE_TDM 1
#endif
#endif
#ifndef USE_TDM
#define USE_TDM 0
#endif

#define LDSADDR(p) ((unsigned)(unsigned long long)(const void*)(p))

union FragAB { v16bf v; uint4 u[2]; };

__device__ __forceinline__ unsigned short f2bf(float f) {
  unsigned u = __float_as_uint(f);
  u += 0x7fffu + ((u >> 16) & 1u);      // round to nearest even
  return (unsigned short)(u >> 16);
}

__device__ __forceinline__ v8f wmma_bf16(const FragAB& a, const FragAB& b, v8f c) {
  return __builtin_amdgcn_wmma_f32_16x16x32_bf16(false, a.v, false, b.v,
                                                 (short)0, c, false, false);
}

// Issue one TDM 2D tile load: tile tw x th (elems), row stride strideElems,
// 2-byte elements, global -> LDS (contiguous th rows of tw elems).
__device__ __forceinline__ void tdm_load_2d(unsigned lds,
                                            const unsigned short* gp,
                                            unsigned strideElems,
                                            unsigned tw, unsigned th) {
#if USE_TDM
  const unsigned long long ga = (unsigned long long)gp;
  // D# group0: count=1 | lds_addr | global_addr[56:0] | type=2
  v4u g0 = {1u, lds, (unsigned)(ga & 0xffffffffu),
            (unsigned)((ga >> 32) & 0x01ffffffu) | (2u << 30)};
  // D# group1: data_size=2B; tensor dims == tile dims (no OOB); dim0 stride
  v8i g1 = {(int)(1u << 16),          // data_size = 1 -> 2-byte elements
            (int)(tw << 16),          // tensor_dim0[15:0]
            (int)(th << 16),          // tensor_dim0[31:16]=0 | tensor_dim1[15:0]
            (int)(tw << 16),          // tensor_dim1[31:16]=0 | tile_dim0
            (int)th,                  // tile_dim1 | tile_dim2=0
            (int)strideElems,         // tensor_dim0_stride[31:0]
            0, 0};
  v4i gz = {0, 0, 0, 0};
#if __clang_major__ >= 23
  v8i gz8 = {0, 0, 0, 0, 0, 0, 0, 0};
  __builtin_amdgcn_tensor_load_to_lds(g0, g1, gz, gz, gz8, 0);
#else
  __builtin_amdgcn_tensor_load_to_lds(g0, g1, gz, gz, 0);
#endif
#endif
}

// ------------------------------ converts -----------------------------------

__global__ void f32_to_bf16_kernel(const float* __restrict__ src,
                                   unsigned short* __restrict__ dst, int n) {
  for (int i = blockIdx.x * blockDim.x + threadIdx.x; i < n;
       i += gridDim.x * blockDim.x)
    dst[i] = f2bf(src[i]);
}

// kv_cache K half [B,H,LC,D] f32 -> ktb [B,H,LT,D] bf16 (prefix l<LC)
__global__ void cache_k_to_bf16_kernel(const float* __restrict__ src,
                                       unsigned short* __restrict__ kt,
                                       int total) {
  for (int i = blockIdx.x * blockDim.x + threadIdx.x; i < total;
       i += gridDim.x * blockDim.x) {
    const int d  = i & 63;
    const int l  = (i >> 6) & 4095;
    const int bh = i >> 18;            // 64*4096 = 2^18
    kt[((size_t)bh * LT_ + l) * D_ + d] = f2bf(src[i]);
  }
}

// Tiled transpose-convert: src f32 [R][Cc] (batched) -> dst bf16 [Cc][R']
// with dst row stride dstStride. 64x64 tiles via LDS, coalesced both sides.
__global__ void __launch_bounds__(128)
transpose_f32_to_bf16_kernel(const float* __restrict__ src,
                             unsigned short* __restrict__ dst,
                             int Cc, int dstStride,
                             long long srcBatch, long long dstBatch) {
  __shared__ unsigned short T[64][72];
  const int r0 = blockIdx.y * 64, c0 = blockIdx.x * 64;
  const float* s = src + (size_t)blockIdx.z * srcBatch;
  unsigned short* d = dst + (size_t)blockIdx.z * dstBatch;
  for (int k = threadIdx.x; k < 4096; k += 128) {
    const int r = k >> 6, c = k & 63;
    T[c][r] = f2bf(s[(size_t)(r0 + r) * Cc + c0 + c]);
  }
  __syncthreads();
  for (int k = threadIdx.x; k < 4096; k += 128) {
    const int c = k >> 6, r = k & 63;
    d[(size_t)(c0 + c) * dstStride + r0 + r] = T[c][r];
  }
}

// ------------------------------ WMMA GEMM ----------------------------------
// C[M,Nn] = A[M,K] * B[K,Nn] with B supplied TRANSPOSED as BT[Nn][K].
// Block tile 128x128, 8 waves (2 M x 4 N), wave tile 64x32, K-step 32.
// Double-buffered LDS; tiles streamed by the TDM while WMMA computes.
// MODE 0: QKV epilogue (split q/k/v, scale q, scatter; V scattered into the
//         transposed [B,H,D,LT] layout). MODE 1: bias + f32 store.
template <int MODE>
__global__ void __launch_bounds__(256)
gemm_bf16_kernel(const unsigned short* __restrict__ A,
                 const unsigned short* __restrict__ BT,
                 int M, int Nn, int K,
                 unsigned short* __restrict__ qout,
                 unsigned short* __restrict__ kout,
                 unsigned short* __restrict__ vout,
                 const float* __restrict__ bias,
                 float* __restrict__ outf) {
  __shared__ unsigned short As[2][128][32];   // [m][k]
  __shared__ unsigned short Bs[2][128][32];   // [n][k]

  const int t = threadIdx.x;
  const int lane = t & 31, wave = t >> 5;
  const int wm = wave >> 2, wn = wave & 3;
  const int hl = lane >> 4, l16 = lane & 15;
  const int base8 = hl * 8, base16 = hl * 16;
  const int bm = blockIdx.y * 128, bn = blockIdx.x * 128;

  const v8f z8 = {0.f, 0.f, 0.f, 0.f, 0.f, 0.f, 0.f, 0.f};
  v8f acc[4][2];
  for (int i = 0; i < 4; i++)
    for (int j = 0; j < 2; j++) acc[i][j] = z8;

  const int NT = K / 32;
#if USE_TDM
  if (wave == 0) {
    tdm_load_2d(LDSADDR(&As[0][0][0]), A  + (size_t)bm * K, K, 32, 128);
    tdm_load_2d(LDSADDR(&Bs[0][0][0]), BT + (size_t)bn * K, K, 32, 128);
  }
#endif
  for (int i = 0; i < NT; i++) {
    const int kk = i * 32, cur = i & 1, nxt = cur ^ 1;
#if USE_TDM
    if (wave == 0) {
      if (i + 1 < NT) {
        tdm_load_2d(LDSADDR(&As[nxt][0][0]), A  + (size_t)bm * K + kk + 32,
                    K, 32, 128);
        tdm_load_2d(LDSADDR(&Bs[nxt][0][0]), BT + (size_t)bn * K + kk + 32,
                    K, 32, 128);
        __builtin_amdgcn_s_wait_tensorcnt(2);   // pair(i) landed (in-order)
      } else {
        __builtin_amdgcn_s_wait_tensorcnt(0);
      }
    }
#else
    {
      const int row = t >> 1, col = (t & 1) * 16;
      const uint4* ga = (const uint4*)(A + (size_t)(bm + row) * K + kk + col);
      *(uint4*)&As[cur][row][col]     = ga[0];
      *(uint4*)&As[cur][row][col + 8] = ga[1];
      const uint4* gb = (const uint4*)(BT + (size_t)(bn + row) * K + kk + col);
      *(uint4*)&Bs[cur][row][col]     = gb[0];
      *(uint4*)&Bs[cur][row][col + 8] = gb[1];
    }
#endif
    __syncthreads();

    FragAB fa[4], fb[2];
#pragma unroll
    for (int mt = 0; mt < 4; mt++) {
      const unsigned short* p = &As[cur][wm * 64 + mt * 16 + l16][0];
      fa[mt].u[0] = *(const uint4*)(p + base16);       // K = base16..+7
      fa[mt].u[1] = *(const uint4*)(p + base16 + 8);   // K = base16+8..+15
    }
#pragma unroll
    for (int nt = 0; nt < 2; nt++) {
      const unsigned short* p = &Bs[cur][wn * 32 + nt * 16 + l16][0];
      fb[nt].u[0] = *(const uint4*)(p + base16);
      fb[nt].u[1] = *(const uint4*)(p + base16 + 8);
    }
#pragma unroll
    for (int mt = 0; mt < 4; mt++)
#pragma unroll
      for (int nt = 0; nt < 2; nt++)
        acc[mt][nt] = wmma_bf16(fa[mt], fb[nt], acc[mt][nt]);
    __syncthreads();   // retire readers before buffer is overwritten
  }

  // epilogue: element r of lane <-> row r + 8*hl, col l16 within 16x16 tile
#pragma unroll
  for (int mt = 0; mt < 4; mt++) {
#pragma unroll
    for (int nt = 0; nt < 2; nt++) {
#pragma unroll
      for (int r = 0; r < 8; r++) {
        const int grow = bm + wm * 64 + mt * 16 + r + hl * 8;
        const int gcol = bn + wn * 32 + nt * 16 + l16;
        const float val = acc[mt][nt][r];
        if (MODE == 0) {
          const int wh = gcol >> 10;      // 0=q 1=k 2=v
          const int c  = gcol & 1023;
          const int h  = c >> 6, dd = c & 63;
          const int b  = grow >> 10, n = grow & 1023;
          const int bh = b * H_ + h;
          if (wh == 0)
            qout[(((size_t)bh) * N_ + n) * D_ + dd] = f2bf(val * 0.125f);
          else if (wh == 1)
            kout[(((size_t)bh) * LT_ + LC_ + n) * D_ + dd] = f2bf(val);
          else  // V goes into transposed [B,H,D,LT] layout
            vout[((size_t)(bh * D_ + dd)) * LT_ + LC_ + n] = f2bf(val);
        } else {
          outf[(size_t)grow * Nn + gcol] = val + bias[gcol];
        }
      }
    }
  }
}

// --------------------------- flash attention -------------------------------
// grid = (N/64, B*H), block = 128 (4 waves). Wave w owns 16 query rows.
// 64-key chunks; K [B,H,LT,D] and transposed V [B,H,D,LT] tiles streamed by
// the TDM into double-buffered LDS while WMMA consumes the previous chunk.
__global__ void __launch_bounds__(128)
flash_attn_kernel(const unsigned short* __restrict__ Q,
                  const unsigned short* __restrict__ Kc,
                  const unsigned short* __restrict__ Vt,
                  unsigned short* __restrict__ O) {
  __shared__ unsigned short Qs[64][64];              // [q][d]
  __shared__ unsigned short Ks[2][CHUNK][64];        // [key][d]
  __shared__ unsigned short Vs[2][64][CHUNK];        // [d][key]
  __shared__ unsigned short Ps[4][16][CHUNK];        // wave-private P tiles

  const int t = threadIdx.x;
  const int lane = t & 31, wave = t >> 5;
  const int hl = lane >> 4, l16 = lane & 15;
  const int base8 = hl * 8, base16 = hl * 16;
  const int bh = blockIdx.y;
  const int qbase = blockIdx.x * 64;
  const size_t qoff = ((size_t)bh * N_ + qbase) * D_;
  const size_t koff = (size_t)bh * LT_ * D_;         // K: [B,H,LT,D]
  const size_t voff = (size_t)bh * D_ * LT_;         // V: [B,H,D,LT]

  // stage Q tile (64x64); visibility covered by first loop barrier
  {
    const int row = t >> 1, half = t & 1;
    const uint4* g = (const uint4*)(Q + qoff + (size_t)row * D_ + half * 32);
    uint4* s = (uint4*)&Qs[row][half * 32];
    s[0] = g[0]; s[1] = g[1]; s[2] = g[2]; s[3] = g[3];
  }

  const v8f z8 = {0.f, 0.f, 0.f, 0.f, 0.f, 0.f, 0.f, 0.f};
  v8f oacc[4];
  for (int i = 0; i < 4; i++) oacc[i] = z8;
  float m_i[8], l_i[8];
#pragma unroll
  for (int r = 0; r < 8; r++) { m_i[r] = -3.0e38f; l_i[r] = 0.f; }

  const int NCH = LT_ / CHUNK;
#if USE_TDM
  if (wave == 0) {
    tdm_load_2d(LDSADDR(&Ks[0][0][0]), Kc + koff, 64, 64, CHUNK);
    tdm_load_2d(LDSADDR(&Vs[0][0][0]), Vt + voff, LT_, CHUNK, 64);
  }
#endif
  for (int i = 0; i < NCH; i++) {
    const int l0 = i * CHUNK;
    const int cur = i & 1, nxt = cur ^ 1;
#if USE_TDM
    if (wave == 0) {
      if (i + 1 < NCH) {
        tdm_load_2d(LDSADDR(&Ks[nxt][0][0]),
                    Kc + koff + (size_t)(l0 + CHUNK) * D_, 64, 64, CHUNK);
        tdm_load_2d(LDSADDR(&Vs[nxt][0][0]),
                    Vt + voff + l0 + CHUNK, LT_, CHUNK, 64);
        __builtin_amdgcn_s_wait_tensorcnt(2);   // chunk i's pair landed
      } else {
        __builtin_amdgcn_s_wait_tensorcnt(0);
      }
    }
#else
    {
      const int row = t >> 1, seg = t & 1;   // row = key for K, d for V
      const uint4* gk =
          (const uint4*)(Kc + koff + (size_t)(l0 + row) * D_ + seg * 32);
      uint4* sk = (uint4*)&Ks[cur][row][seg * 32];
      sk[0] = gk[0]; sk[1] = gk[1]; sk[2] = gk[2]; sk[3] = gk[3];
      const uint4* gv =
          (const uint4*)(Vt + voff + (size_t)row * LT_ + l0 + seg * 32);
      uint4* sv = (uint4*)&Vs[cur][row][seg * 32];
      sv[0] = gv[0]; sv[1] = gv[1]; sv[2] = gv[2]; sv[3] = gv[3];
    }
#endif
    if (l0 + 2 * CHUNK < LT_)   // global_prefetch_b8 two chunks ahead
      __builtin_prefetch(Kc + koff + (size_t)(l0 + 2 * CHUNK) * D_ + t * 64, 0, 1);
    __syncthreads();

    // S = Q * K^T : wave's 16 rows x 64 keys
    FragAB fq[2], fk;
#pragma unroll
    for (int ks = 0; ks < 2; ks++) {
      const unsigned short* p = &Qs[wave * 16 + l16][ks * 32];
      fq[ks].u[0] = *(const uint4*)(p + base16);
      fq[ks].u[1] = *(const uint4*)(p + base16 + 8);
    }
    v8f sf[4];
#pragma unroll
    for (int nt = 0; nt < 4; nt++) {
      sf[nt] = z8;
#pragma unroll
      for (int ks = 0; ks < 2; ks++) {
        const unsigned short* p = &Ks[cur][nt * 16 + l16][ks * 32 + base16];
        fk.u[0] = *(const uint4*)(p);
        fk.u[1] = *(const uint4*)(p + 8);
        sf[nt] = wmma_bf16(fq[ks], fk, sf[nt]);
      }
    }

    // online softmax per element-row r (row = r + 8*hl)
#pragma unroll
    for (int r = 0; r < 8; r++) {
      float mx = fmaxf(fmaxf(sf[0][r], sf[1][r]), fmaxf(sf[2][r], sf[3][r]));
#pragma unroll
      for (int mask = 1; mask < 16; mask <<= 1)
        mx = fmaxf(mx, __shfl_xor(mx, mask, 32));
      const float m_new = fmaxf(m_i[r], mx);
      const float alpha = __expf(m_i[r] - m_new);
      float p[4], rs = 0.f;
#pragma unroll
      for (int nt = 0; nt < 4; nt++) {
        p[nt] = __expf(sf[nt][r] - m_new);
        rs += p[nt];
      }
#pragma unroll
      for (int mask = 1; mask < 16; mask <<= 1) rs += __shfl_xor(rs, mask, 32);
      l_i[r] = l_i[r] * alpha + rs;
      m_i[r] = m_new;
#pragma unroll
      for (int nt = 0; nt < 4; nt++) oacc[nt][r] *= alpha;
#pragma unroll
      for (int nt = 0; nt < 4; nt++)
        Ps[wave][r + hl * 8][l16 + nt * 16] = f2bf(p[nt]);
    }
    // wave-private LDS transpose: only this wave's own DS ops must land
    asm volatile("s_wait_dscnt 0x0" ::: "memory");

    // O += P * V  (contract over 64 keys = two 32-key WMMA steps)
    FragAB fp, fv;
#pragma unroll
    for (int ks = 0; ks < 2; ks++) {
      const unsigned short* pp = &Ps[wave][l16][ks * 32];
      fp.u[0] = *(const uint4*)(pp + base16);
      fp.u[1] = *(const uint4*)(pp + base16 + 8);
#pragma unroll
      for (int nt = 0; nt < 4; nt++) {
        const unsigned short* pv = &Vs[cur][nt * 16 + l16][ks * 32 + base16];
        fv.u[0] = *(const uint4*)(pv);
        fv.u[1] = *(const uint4*)(pv + 8);
        oacc[nt] = wmma_bf16(fp, fv, oacc[nt]);
      }
    }
    __syncthreads();   // retire readers before this buffer is re-filled
  }

  // write O[b, n, h*64 + d] as bf16 (feeds proj GEMM)
  const int b = bh >> 4, h = bh & 15;
#pragma unroll
  for (int r = 0; r < 8; r++) {
    const float inv = 1.f / l_i[r];
    const int n = qbase + wave * 16 + r + hl * 8;
#pragma unroll
    for (int nt = 0; nt < 4; nt++) {
      const int d = nt * 16 + l16;
      O[((size_t)b * N_ + n) * C_ + h * 64 + d] = f2bf(oacc[nt][r] * inv);
    }
  }
}

// ------------------------------ launcher -----------------------------------

extern "C" void kernel_launch(void* const* d_in, const int* in_sizes, int n_in,
                              void* d_out, int out_size, void* d_ws,
                              size_t ws_size, hipStream_t stream) {
  const float* x     = (const float*)d_in[0];   // [B,N,C]
  const float* kvc   = (const float*)d_in[1];   // [2,B,H,LC,D]
  const float* wqkv  = (const float*)d_in[2];   // [C,3C]
  const float* wproj = (const float*)d_in[3];   // [C,C]
  const float* bproj = (const float*)d_in[4];   // [C]
  float* out = (float*)d_out;                   // [B,N,C] f32

  // workspace layout (bf16 buffers), ~112 MB total
  char* ws = (char*)d_ws;
  const size_t SZ_XB    = (size_t)B_ * N_ * C_ * 2;        // 8 MB
  const size_t SZ_WQKV  = (size_t)C_ * 3 * C_ * 2;         // 6 MB  (stored [3C][C])
  const size_t SZ_WPROJ = (size_t)C_ * C_ * 2;             // 2 MB  (stored [C][C] T)
  const size_t SZ_Q     = (size_t)B_ * H_ * N_ * D_ * 2;   // 8 MB
  const size_t SZ_KT    = (size_t)B_ * H_ * LT_ * D_ * 2;  // 40 MB
  unsigned short* xb    = (unsigned short*)(ws);
  unsigned short* wqkvT = (unsigned short*)(ws + SZ_XB);
  unsigned short* wprjT = (unsigned short*)(ws + SZ_XB + SZ_WQKV);
  unsigned short* qb    = (unsigned short*)(ws + SZ_XB + SZ_WQKV + SZ_WPROJ);
  unsigned short* ktb   = (unsigned short*)(ws + SZ_XB + SZ_WQKV + SZ_WPROJ + SZ_Q);
  unsigned short* vtb   = (unsigned short*)(ws + SZ_XB + SZ_WQKV + SZ_WPROJ + SZ_Q + SZ_KT);
  unsigned short* aob   = (unsigned short*)(ws + SZ_XB + SZ_WQKV + SZ_WPROJ + SZ_Q + 2 * SZ_KT);

  // 1) converts / transposes
  f32_to_bf16_kernel<<<2048, 256, 0, stream>>>(x, xb, B_ * N_ * C_);
  // w_qkv [1024][3072] -> wqkvT [3072][1024]
  transpose_f32_to_bf16_kernel<<<dim3(48, 16, 1), 128, 0, stream>>>(
      wqkv, wqkvT, 3 * C_, C_, 0, 0);
  // w_proj [1024][1024] -> wprjT [1024][1024]
  transpose_f32_to_bf16_kernel<<<dim3(16, 16, 1), 128, 0, stream>>>(
      wproj, wprjT, C_, C_, 0, 0);
  // K cache: linear convert into [B,H,LT,D] prefix
  cache_k_to_bf16_kernel<<<4096, 256, 0, stream>>>(kvc, ktb,
                                                   B_ * H_ * LC_ * D_);
  // V cache: per-(b,h) transpose [LC,D] -> [D,LT] prefix
  transpose_f32_to_bf16_kernel<<<dim3(1, 64, 64), 128, 0, stream>>>(
      kvc + (size_t)B_ * H_ * LC_ * D_, vtb, D_, LT_,
      (long long)LC_ * D_, (long long)D_ * LT_);

  // 2) QKV GEMM: [4096,1024] x [1024,3072], fused split/scale/scatter
  gemm_bf16_kernel<0><<<dim3((3 * C_) / 128, (B_ * N_) / 128), 256, 0, stream>>>(
      xb, wqkvT, B_ * N_, 3 * C_, C_, qb, ktb, vtb, nullptr, nullptr);

  // 3) flash attention over L_total = 5120
  flash_attn_kernel<<<dim3(N_ / 64, B_ * H_), 128, 0, stream>>>(qb, ktb, vtb, aob);

  // 4) proj GEMM: [4096,1024] x [1024,1024] + bias -> f32 out
  gemm_bf16_kernel<1><<<dim3(C_ / 128, (B_ * N_) / 128), 256, 0, stream>>>(
      aob, wprjT, B_ * N_, C_, C_, nullptr, nullptr, nullptr, bproj, out);
}
